// CFI_Module_21260088115731
// MI455X (gfx1250) — matrix operand order repaired and checked by reading, not compiled, not run on hardware
//
#include <hip/hip_runtime.h>
#include <hip/hip_bf16.h>
#include <math.h>

typedef _Float16 h16;
typedef __attribute__((ext_vector_type(16))) _Float16 v16h;
typedef __attribute__((ext_vector_type(8)))  _Float16 v8h;
typedef __attribute__((ext_vector_type(8)))  float    v8f;
typedef __attribute__((ext_vector_type(4)))  float    v4f;

// Problem constants (b=4, c=256, h=w=64 -> N=2048 after the channel re-view)
#define NB    4
#define HW    4096     // h*w
#define SEQ   2048     // N = h*w/2
#define CC    256      // feature dim of the attention (c)
#define KALL  640      // 512 AB channels + 128 y channels
#define OUTC  256

#define WMMA16(a, b, c) \
  __builtin_amdgcn_wmma_f32_16x16x32_f16(false, (a), false, (b), (short)0, (c), false, false)

// -------- WMMA fragment loaders (CDNA5 16-bit 16x32 A / 32x16 B layouts) --------
// A (16x32 f16): lane = {row = lane&15, half = lane>>4}; lane holds
//   K = half*8 + [0..7]  (frag elts 0..7 / VGPR0-3) and 16+half*8+[0..7] (elts 8..15).
__device__ __forceinline__ v16h frag_a_rowK(const h16* __restrict__ base, int ld,
                                            int kbase, int lane) {
  const int r = lane & 15, h = lane >> 4;
  const h16* p = base + (size_t)r * ld + kbase + h * 8;
  v8h lo = *(const v8h*)(p);
  v8h hi = *(const v8h*)(p + 16);
  v16h f;
#pragma unroll
  for (int i = 0; i < 8; ++i) { f[i] = lo[i]; f[8 + i] = hi[i]; }
  return f;
}

// B (32x16 f16) fed from [col][K] storage: lane = {col = lane&15, half = lane>>4};
// lane holds K = half*8 + [0..7] (elts 0..7) and 16+half*8+[0..7] (elts 8..15).
__device__ __forceinline__ v16h frag_b_colK(const h16* __restrict__ base, int ld,
                                            int kbase, int lane) {
  const int c = lane & 15, h = lane >> 4;
  const h16* p = base + (size_t)c * ld + kbase + h * 8;
  v8h lo = *(const v8h*)(p);
  v8h hi = *(const v8h*)(p + 16);
  v16h f;
#pragma unroll
  for (int i = 0; i < 8; ++i) { f[i] = lo[i]; f[8 + i] = hi[i]; }
  return f;
}

// Store one 16x16 fp32 D tile: row index = N-coord (lane&15), col index = M-coord
// (two contiguous float4 per lane).
__device__ __forceinline__ void store_d_f32(const v8f& acc, float* __restrict__ base,
                                            int ld, int lane) {
  float* o = base + (size_t)(lane & 15) * ld + (lane >> 4) * 8;
  v4f lo = {acc[0], acc[1], acc[2], acc[3]};
  v4f hi = {acc[4], acc[5], acc[6], acc[7]};
  *(v4f*)(o)     = lo;
  *(v4f*)(o + 4) = hi;
}

// -------- 1) weights -> f16 (and build concatenated [W_AB | W_mask] 256x640) ------
__global__ void k_prep_weights(const float* __restrict__ Wphi, const float* __restrict__ Wth,
                               const float* __restrict__ Wg,  const float* __restrict__ WAB,
                               const float* __restrict__ Wmask,
                               h16* __restrict__ WphiH, h16* __restrict__ WthH,
                               h16* __restrict__ WgH,   h16* __restrict__ WallH) {
  int i = blockIdx.x * blockDim.x + threadIdx.x;
  if (i < 32768) {                     // W_phi 128x256
    WphiH[i] = (h16)Wphi[i];
  } else if (i < 65536) {              // W_theta 128x256
    int t = i - 32768; WthH[t] = (h16)Wth[t];
  } else if (i < 131072) {             // W_g 128x512
    int t = i - 65536; WgH[t] = (h16)Wg[t];
  } else if (i < 131072 + OUTC * KALL) {
    int t = i - 131072; int o = t / KALL, k = t - o * KALL;
    WallH[t] = (h16)((k < 512) ? WAB[o * 512 + k] : Wmask[o * 128 + (k - 512)]);
  }
}

// -------- 2) LDS-tiled transpose: XallT[b][p][ch] = {A|B}[b][ch][p] (f32 -> f16) ----
__global__ void k_transpose(const float* __restrict__ A, const float* __restrict__ B,
                            h16* __restrict__ XallT) {
  __shared__ float tile[32][33];
  const int b  = blockIdx.z;
  const int p0 = blockIdx.x * 32, c0 = blockIdx.y * 32;   // c0 over 512 combined channels
  const int tx = threadIdx.x, ty = threadIdx.y;           // block (32, 8)
  const float* src = (c0 < 256)
      ? (A + (size_t)b * 256 * HW + (size_t)c0 * HW)
      : (B + (size_t)b * 256 * HW + (size_t)(c0 - 256) * HW);
#pragma unroll
  for (int j = 0; j < 32; j += 8) tile[ty + j][tx] = src[(size_t)(ty + j) * HW + p0 + tx];
  __syncthreads();
  h16* dst = XallT + (size_t)b * HW * KALL;
#pragma unroll
  for (int j = 0; j < 32; j += 8)
    dst[(size_t)(p0 + ty + j) * KALL + c0 + tx] = (h16)tile[tx][ty + j];
}

// -------- 3) projection GEMMs: proj[j][p] = sum_k W[j][k] * Xall[kofs+k][p] ---------
// 32x32 macro-tile per wave: M=p (rows from XallT), N=j (weight rows).
// Store with the channel re-view: (j,p) -> cc = 2j + (p>=2048), row = p&2047.
// mode 0: dst[row][cc] (2048x256, phiRM/thetaRM); mode 1: dst[cc][row] (256x2048, gT)
__global__ void k_proj(const h16* __restrict__ XallT, const h16* __restrict__ W,
                       h16* __restrict__ dst, int kofs, int Kw, int mode) {
  const int wid  = (blockIdx.x * blockDim.x + threadIdx.x) >> 5;
  const int lane = threadIdx.x & 31;
  const int b  = wid >> 9;            // 512 macro-tiles per batch
  const int t  = wid & 511;
  const int pt = t >> 2, jt = t & 3;  // 128 p-tiles x 4 j-tiles (32 wide each)
  const h16* Abase = XallT + (size_t)b * HW * KALL + (size_t)(pt * 32) * KALL;
  const h16* Bbase = W + (size_t)(jt * 32) * Kw;
  v8f acc00 = {}, acc01 = {}, acc10 = {}, acc11 = {};
  for (int k = 0; k < Kw; k += 32) {
    if (k + 32 < Kw) {
      __builtin_prefetch(Abase + (size_t)(lane & 15) * KALL + kofs + k + 32, 0, 1);
      __builtin_prefetch(Bbase + (size_t)(lane & 15) * Kw + k + 32, 0, 1);
    }
    v16h a0 = frag_a_rowK(Abase, KALL, kofs + k, lane);
    v16h a1 = frag_a_rowK(Abase + (size_t)16 * KALL, KALL, kofs + k, lane);
    v16h w0 = frag_b_colK(Bbase, Kw, k, lane);
    v16h w1 = frag_b_colK(Bbase + (size_t)16 * Kw, Kw, k, lane);
    acc00 = WMMA16(a0, w0, acc00);
    acc01 = WMMA16(a0, w1, acc01);
    acc10 = WMMA16(a1, w0, acc10);
    acc11 = WMMA16(a1, w1, acc11);
  }
  const v8f accs[4] = {acc00, acc01, acc10, acc11};
#pragma unroll
  for (int mi = 0; mi < 2; ++mi)
#pragma unroll
    for (int ni = 0; ni < 2; ++ni) {
      const v8f& a = accs[mi * 2 + ni];
      const int j = jt * 32 + ni * 16 + (lane & 15);
      const int pbase = pt * 32 + mi * 16 + (lane >> 4) * 8;
#pragma unroll
      for (int i = 0; i < 8; ++i) {
        const int p = pbase + i;
        const int cc = 2 * j + (p >> 11);
        const int row = p & 2047;
        if (mode == 0) dst[(size_t)b * SEQ * CC + (size_t)row * CC + cc] = (h16)a[i];
        else           dst[(size_t)b * CC * SEQ + (size_t)cc * SEQ + row] = (h16)a[i];
      }
    }
}

// -------- 4) scores: S[n][m] = sum_cc theta[n][cc] * phi[cc][m] ---------------------
// 32x32 macro-tile: M=m (rows from phiRM), N=n (cols from thetaRM).
__global__ void k_gemm_scores(const h16* __restrict__ phiRM, const h16* __restrict__ thetaRM,
                              float* __restrict__ S) {
  const int wid  = (blockIdx.x * blockDim.x + threadIdx.x) >> 5;
  const int lane = threadIdx.x & 31;
  const int b  = wid >> 12;            // 64x64 macro-tiles per batch
  const int t  = wid & 4095;
  const int mt = t >> 6, nt = t & 63;
  const h16* Abase = phiRM   + (size_t)b * SEQ * CC + (size_t)(mt * 32) * CC;
  const h16* Bbase = thetaRM + (size_t)b * SEQ * CC + (size_t)(nt * 32) * CC;
  v8f acc00 = {}, acc01 = {}, acc10 = {}, acc11 = {};
#pragma unroll
  for (int k = 0; k < CC; k += 32) {
    if (k + 32 < CC) {
      __builtin_prefetch(Abase + (size_t)(lane & 15) * CC + k + 32, 0, 1);
      __builtin_prefetch(Bbase + (size_t)(lane & 15) * CC + k + 32, 0, 1);
    }
    v16h a0 = frag_a_rowK(Abase, CC, k, lane);
    v16h a1 = frag_a_rowK(Abase + (size_t)16 * CC, CC, k, lane);
    v16h b0 = frag_b_colK(Bbase, CC, k, lane);
    v16h b1 = frag_b_colK(Bbase + (size_t)16 * CC, CC, k, lane);
    acc00 = WMMA16(a0, b0, acc00);
    acc01 = WMMA16(a0, b1, acc01);
    acc10 = WMMA16(a1, b0, acc10);
    acc11 = WMMA16(a1, b1, acc11);
  }
  float* Sb = S + (size_t)b * SEQ * SEQ;
  // tile(mi, ni): rows n = nt*32+ni*16.., cols m = mt*32+mi*16..
  store_d_f32(acc00, Sb + (size_t)(nt * 32 +  0) * SEQ + mt * 32 +  0, SEQ, lane);
  store_d_f32(acc01, Sb + (size_t)(nt * 32 + 16) * SEQ + mt * 32 +  0, SEQ, lane);
  store_d_f32(acc10, Sb + (size_t)(nt * 32 +  0) * SEQ + mt * 32 + 16, SEQ, lane);
  store_d_f32(acc11, Sb + (size_t)(nt * 32 + 16) * SEQ + mt * 32 + 16, SEQ, lane);
}

// -------- 5) column softmax stats: max_n S[n][m], invZ[m] = 1/sum_n exp(S-max) ------
__global__ void k_colstats(const float* __restrict__ S, float* __restrict__ colmax,
                           float* __restrict__ invZ) {
  const int idx = blockIdx.x * blockDim.x + threadIdx.x;   // 0..NB*SEQ-1
  const int b = idx >> 11, m = idx & 2047;
  const float* col = S + (size_t)b * SEQ * SEQ + m;
  float mx = -INFINITY;
  for (int n = 0; n < SEQ; ++n) mx = fmaxf(mx, col[(size_t)n * SEQ]);
  float s = 0.f;
  for (int n = 0; n < SEQ; ++n) s += __expf(col[(size_t)n * SEQ] - mx);
  colmax[idx] = mx;
  invZ[idx]   = 1.0f / s;
}

// Build one A fragment of P[n][m] = exp(S[n][m]-max[m]) * invZ[m] on the fly.
__device__ __forceinline__ v16h frag_p(const float* __restrict__ srow,
                                       const float* __restrict__ mxb,
                                       const float* __restrict__ izb,
                                       int m0, int h) {
  v16h a;
#pragma unroll
  for (int g = 0; g < 2; ++g) {
    const int kb = m0 + g * 16 + h * 8;
    v4f s0 = *(const v4f*)(srow + kb);
    v4f s1 = *(const v4f*)(srow + kb + 4);
    v4f x0 = *(const v4f*)(mxb + kb);
    v4f x1 = *(const v4f*)(mxb + kb + 4);
    v4f z0 = *(const v4f*)(izb + kb);
    v4f z1 = *(const v4f*)(izb + kb + 4);
#pragma unroll
    for (int i = 0; i < 4; ++i) {
      a[g * 8 + i]     = (h16)(__expf(s0[i] - x0[i]) * z0[i]);
      a[g * 8 + 4 + i] = (h16)(__expf(s1[i] - x1[i]) * z1[i]);
    }
  }
  return a;
}

// -------- 6) y = P * g, 32x32 macro-tile: M=n (exp of S rows), N=cc (gT [cc][m]) ----
// Output written as f16 straight into XallT channels 512..639 via the re-view.
__global__ void k_gemm_y(const float* __restrict__ S, const float* __restrict__ colmax,
                         const float* __restrict__ invZ, const h16* __restrict__ gT,
                         h16* __restrict__ XallT) {
  const int wid  = (blockIdx.x * blockDim.x + threadIdx.x) >> 5;
  const int lane = threadIdx.x & 31;
  const int b  = wid >> 9;             // 64 n-tiles x 8 cc-tiles (32 wide) per batch
  const int t  = wid & 511;
  const int nt = t >> 3, ct = t & 7;
  const int r = lane & 15, h = lane >> 4;
  const float* srow0 = S + (size_t)b * SEQ * SEQ + (size_t)(nt * 32 + r) * SEQ;
  const float* srow1 = srow0 + (size_t)16 * SEQ;
  const float* mxb = colmax + (size_t)b * SEQ;
  const float* izb = invZ   + (size_t)b * SEQ;
  const h16* Bbase = gT + (size_t)b * CC * SEQ + (size_t)(ct * 32) * SEQ;
  v8f acc00 = {}, acc01 = {}, acc10 = {}, acc11 = {};
  for (int m0 = 0; m0 < SEQ; m0 += 32) {
    if (m0 + 32 < SEQ) {
      __builtin_prefetch(srow0 + m0 + 32, 0, 1);
      __builtin_prefetch(srow1 + m0 + 32, 0, 1);
      __builtin_prefetch(Bbase + (size_t)r * SEQ + m0 + 32, 0, 1);
    }
    v16h a0 = frag_p(srow0, mxb, izb, m0, h);
    v16h a1 = frag_p(srow1, mxb, izb, m0, h);
    v16h b0 = frag_b_colK(Bbase, SEQ, m0, lane);
    v16h b1 = frag_b_colK(Bbase + (size_t)16 * SEQ, SEQ, m0, lane);
    acc00 = WMMA16(a0, b0, acc00);
    acc01 = WMMA16(a0, b1, acc01);
    acc10 = WMMA16(a1, b0, acc10);
    acc11 = WMMA16(a1, b1, acc11);
  }
  // (n, cc) -> XallT[p = (cc&1)*2048 + n][512 + (cc>>1)]
  h16* dst = XallT + (size_t)b * HW * KALL;
  const v8f accs[4] = {acc00, acc01, acc10, acc11};
#pragma unroll
  for (int mi = 0; mi < 2; ++mi)
#pragma unroll
    for (int ni = 0; ni < 2; ++ni) {
      const v8f& a = accs[mi * 2 + ni];
      const int cc = ct * 32 + ni * 16 + (lane & 15);
      const int ch = cc >> 1, hp = cc & 1;
#pragma unroll
      for (int i = 0; i < 8; ++i) {
        const int n = nt * 32 + mi * 16 + h * 8 + i;
        const int p = hp * 2048 + n;
        dst[(size_t)p * KALL + 512 + ch] = (h16)a[i];
      }
    }
}

// -------- 7) out[o][p] = sum_k Wall[o][k] * Xall[k][p]  (fused W_AB + W_mask) ------
// 32x32 macro-tile: M=p (A = XallT rows), N=o (B = Wall rows).
__global__ void k_gemm_out(const h16* __restrict__ XallT, const h16* __restrict__ WallH,
                           float* __restrict__ out) {
  const int wid  = (blockIdx.x * blockDim.x + threadIdx.x) >> 5;
  const int lane = threadIdx.x & 31;
  const int b  = wid >> 10;            // 128 p-tiles x 8 o-tiles (32 wide) per batch
  const int t  = wid & 1023;
  const int pt = t >> 3, ot = t & 7;
  const h16* Abase = XallT + (size_t)b * HW * KALL + (size_t)(pt * 32) * KALL;
  const h16* Bbase = WallH + (size_t)(ot * 32) * KALL;
  v8f acc00 = {}, acc01 = {}, acc10 = {}, acc11 = {};
  for (int k = 0; k < KALL; k += 32) {
    if (k + 32 < KALL) {
      __builtin_prefetch(Abase + (size_t)(lane & 15) * KALL + k + 32, 0, 1);
      __builtin_prefetch(Bbase + (size_t)(lane & 15) * KALL + k + 32, 0, 1);
    }
    v16h a0 = frag_a_rowK(Abase, KALL, k, lane);
    v16h a1 = frag_a_rowK(Abase + (size_t)16 * KALL, KALL, k, lane);
    v16h w0 = frag_b_colK(Bbase, KALL, k, lane);
    v16h w1 = frag_b_colK(Bbase + (size_t)16 * KALL, KALL, k, lane);
    acc00 = WMMA16(a0, w0, acc00);
    acc01 = WMMA16(a0, w1, acc01);
    acc10 = WMMA16(a1, w0, acc10);
    acc11 = WMMA16(a1, w1, acc11);
  }
  float* ob = out + (size_t)b * OUTC * HW;
  // tile(mi, ni): rows o = ot*32+ni*16.., cols p = pt*32+mi*16..
  store_d_f32(acc00, ob + (size_t)(ot * 32 +  0) * HW + pt * 32 +  0, HW, lane);
  store_d_f32(acc01, ob + (size_t)(ot * 32 + 16) * HW + pt * 32 +  0, HW, lane);
  store_d_f32(acc10, ob + (size_t)(ot * 32 +  0) * HW + pt * 32 + 16, HW, lane);
  store_d_f32(acc11, ob + (size_t)(ot * 32 + 16) * HW + pt * 32 + 16, HW, lane);
}

extern "C" void kernel_launch(void* const* d_in, const int* in_sizes, int n_in,
                              void* d_out, int out_size, void* d_ws, size_t ws_size,
                              hipStream_t stream) {
  (void)in_sizes; (void)n_in; (void)out_size; (void)ws_size;
  const float* A     = (const float*)d_in[0];
  const float* B     = (const float*)d_in[1];
  const float* Wphi  = (const float*)d_in[2];
  const float* Wth   = (const float*)d_in[3];
  const float* Wg    = (const float*)d_in[4];
  const float* WAB   = (const float*)d_in[5];
  const float* Wmask = (const float*)d_in[6];
  float* out = (float*)d_out;

  char* ws = (char*)d_ws;
  size_t off = 0;
  auto alloc = [&](size_t bytes) {
    char* p = ws + off;
    off += (bytes + 255) & ~(size_t)255;
    return p;
  };
  h16*   WphiH   = (h16*)alloc((size_t)128 * 256 * 2);
  h16*   WthH    = (h16*)alloc((size_t)128 * 256 * 2);
  h16*   WgH     = (h16*)alloc((size_t)128 * 512 * 2);
  h16*   WallH   = (h16*)alloc((size_t)OUTC * KALL * 2);
  h16*   XallT   = (h16*)alloc((size_t)NB * HW * KALL * 2);     // 20 MB
  h16*   thetaRM = (h16*)alloc((size_t)NB * SEQ * CC * 2);      //  4 MB
  h16*   phiRM   = (h16*)alloc((size_t)NB * SEQ * CC * 2);      //  4 MB
  h16*   gT      = (h16*)alloc((size_t)NB * CC * SEQ * 2);      //  4 MB
  float* S       = (float*)alloc((size_t)NB * SEQ * SEQ * 4);   // 67 MB (fits L2)
  float* colmax  = (float*)alloc((size_t)NB * SEQ * 4);
  float* invZ    = (float*)alloc((size_t)NB * SEQ * 4);

  // 1) weights -> f16
  k_prep_weights<<<(294912 + 255) / 256, 256, 0, stream>>>(
      Wphi, Wth, Wg, WAB, Wmask, WphiH, WthH, WgH, WallH);

  // 2) [A|B] -> XallT (f16, pixel-major)
  {
    dim3 g(HW / 32, 512 / 32, NB);
    dim3 t(32, 8);
    k_transpose<<<g, t, 0, stream>>>(A, B, XallT);
  }

  // 3) projections (phi from A-channels, theta from B-channels, g from both)
  //    NB * 512 waves each -> 256 blocks of 8 waves
  k_proj<<<256, 256, 0, stream>>>(XallT, WphiH, phiRM,   0, 256, 0);
  k_proj<<<256, 256, 0, stream>>>(XallT, WthH,  thetaRM, 256, 256, 0);
  k_proj<<<256, 256, 0, stream>>>(XallT, WgH,   gT,      0, 512, 1);

  // 4) S = theta . phi   (NB * 64 * 64 macro-tiles -> 16384 waves)
  k_gemm_scores<<<2048, 256, 0, stream>>>(phiRM, thetaRM, S);

  // 5) per-column softmax stats
  k_colstats<<<(NB * SEQ) / 256, 256, 0, stream>>>(S, colmax, invZ);

  // 6) y = softmax-normalized(S) . g   -> XallT channels 512..639 (2048 waves)
  k_gemm_y<<<256, 256, 0, stream>>>(S, colmax, invZ, gT, XallT);

  // 7) out = W_mask.y + W_AB.[A|B]   (4096 waves)
  k_gemm_out<<<512, 256, 0, stream>>>(XallT, WallH, out);
}
